// GraphAttentionWithEdgeConcat_70205535420828
// MI455X (gfx1250) — matrix operand order repaired and checked by reading, not compiled, not run on hardware
//
#include <hip/hip_runtime.h>

// ---------------------------------------------------------------------------
// GraphAttentionWithEdgeConcat for MI455X (gfx1250), wave32 + WMMA bf16 path.
// N=4096 nodes, F=256 features, E=3 edge types, OUT=256.
//
// - All heavy GEMMs use v_wmma_f32_16x16x32_bf16 (f32 accumulate).
// - Softmax kernels emit bf16 attention/adj2 matrices directly, halving the
//   A-side traffic of the three K=4096 GEMMs and letting both GEMM operands
//   be staged with GLOBAL_LOAD_ASYNC_TO_LDS_B128 (ASYNCcnt DMA path).
// - K loop is double-buffered (ping-pong LDS, one barrier per K-step), so the
//   next tile's global->LDS DMA overlaps the current tile's WMMAs.
// ---------------------------------------------------------------------------

typedef __attribute__((ext_vector_type(16))) __bf16 v16bf;
typedef __attribute__((ext_vector_type(8)))  __bf16 v8bf;
typedef __attribute__((ext_vector_type(8)))  float  v8f;
typedef int v4i_gcc __attribute__((vector_size(4 * sizeof(int))));  // builtin's operand type

static constexpr int NN  = 4096;   // nodes
static constexpr int FD  = 256;    // features
static constexpr float LEAKY_SLOPE = 0.01f;
static constexpr float NEG_GAT = -9.0e15f;
static constexpr float NEG_MHA = -1.0e9f;

static __device__ __forceinline__ __bf16 f2bf(float x) { return (__bf16)x; }

// ----- async global->LDS 16B copy (falls back to manual copy if the builtin
//       is not available on this toolchain) --------------------------------
#if defined(__has_builtin)
#if __has_builtin(__builtin_amdgcn_global_load_async_to_lds_b128)
#define ASYNC_LDS 1
#endif
#endif
#ifndef ASYNC_LDS
#define ASYNC_LDS 0
#endif

static __device__ __forceinline__ void copy16_g2l(const __bf16* g, __bf16* l)
{
#if ASYNC_LDS
    __builtin_amdgcn_global_load_async_to_lds_b128(
        (__attribute__((address_space(1))) v4i_gcc*)g,
        (__attribute__((address_space(3))) v4i_gcc*)l, 0, 0);
#else
    *(uint4*)l = *(const uint4*)g;
#endif
}

static __device__ __forceinline__ void wait_async()
{
#if ASYNC_LDS
#if __has_builtin(__builtin_amdgcn_s_wait_asynccnt)
    __builtin_amdgcn_s_wait_asynccnt(0);
#else
    asm volatile("s_wait_asynccnt 0" ::: "memory");
#endif
#endif
}

// ---------------------------------------------------------------------------
// Generic tiled GEMM:  C[M x Nc] = A[M x K] * B[K x Nc], BT[Nc x K] in bf16.
// A is f32 (converted during staging) when ABF==false, bf16 when ABF==true.
// Block tile 128x128, K-step 32, 256 threads = 8 waves, each wave 32x64
// (2x4 fragments of 16x16). LDS tiles are stored pre-swizzled into the exact
// WMMA operand-fragment layout so each lane reads its v16bf as 32 contiguous
// bytes (2x ds_load_b128), and each staging store is one 16B LDS write.
//
// MODE: 0 = C = acc
//       1 = C = acc + Res
//       2 = C = relu(acc) + Res
//       3 = C += acc                        (accumulate)
//       4 = C = (Adj!=0) ? acc : NEG_MHA    (masked attention scores)
// Requires M%128==0, Nc%128==0, K%32==0 (true for all calls here).
// ---------------------------------------------------------------------------
template <int MODE, bool ABF>
__global__ __launch_bounds__(256)
void gemm_bf16(const void* __restrict__ Ap, const __bf16* __restrict__ BT,
               float* __restrict__ C, const float* __restrict__ Res,
               const float* __restrict__ Adj,
               int M, int Nc, int K)
{
    __shared__ __bf16 As[2][8 * 512];   // ping-pong 8 A-fragments (16x32), 8 KB each
    __shared__ __bf16 Bs[2][8 * 512];   // ping-pong 8 B-fragments (32x16), 8 KB each

    const int tid  = threadIdx.x;
    const int lane = tid & 31;
    const int wave = tid >> 5;
    const int wm   = wave & 3;        // M quadrant  (wm*32)
    const int wn   = wave >> 2;       // N half      (wn*64)
    const int m0   = blockIdx.x * 128;
    const int n0   = blockIdx.y * 128;

    v8f acc[2][4];
    const v8f vz = {0.f, 0.f, 0.f, 0.f, 0.f, 0.f, 0.f, 0.f};
    #pragma unroll
    for (int i = 0; i < 2; ++i)
        #pragma unroll
        for (int j = 0; j < 4; ++j) acc[i][j] = vz;

    // f32-A staging indices: 128 rows x 32 k, 8 f32 per thread-slot, 2 slots.
    const int arow0 = tid >> 2;       // 0..63 (second pass +64)
    const int akg   = tid & 3;        // k-octet 0..3
    const int algrp = 16 * (akg & 1);
    const int ae0   = 8 * (akg >> 1);
    // bf16-A / B staging indices: 128 rows x 32 k of bf16, 16 bf16 per thread.
    const int srow  = tid >> 1;       // 0..127
    const int skh   = tid & 1;        // k-half

    auto stage = [&](int buf, int k0) {
        if constexpr (!ABF) {
            const float* A = (const float*)Ap;
            #pragma unroll
            for (int r = 0; r < 2; ++r) {
                const int row = arow0 + r * 64;
                const float4* ap = (const float4*)(A + (size_t)(m0 + row) * K + k0 + akg * 8);
                const float4 p0 = ap[0];
                const float4 p1 = ap[1];
                v8bf h;
                h[0] = f2bf(p0.x); h[1] = f2bf(p0.y); h[2] = f2bf(p0.z); h[3] = f2bf(p0.w);
                h[4] = f2bf(p1.x); h[5] = f2bf(p1.y); h[6] = f2bf(p1.z); h[7] = f2bf(p1.w);
                const int fr = row >> 4;
                const int ln = (row & 15) + algrp;
                *(v8bf*)&As[buf][fr * 512 + ln * 16 + ae0] = h;
            }
        } else {
            const __bf16* A = (const __bf16*)Ap;
            #pragma unroll
            for (int o2 = 0; o2 < 2; ++o2) {
                const int o = skh * 2 + o2;            // k-octet 0..3
                const __bf16* src = A + (size_t)(m0 + srow) * K + k0 + o * 8;
                __bf16* dst = &As[buf][(srow >> 4) * 512 +
                                       ((srow & 15) + 16 * (o & 1)) * 16 + 8 * (o >> 1)];
                copy16_g2l(src, dst);
            }
        }
        {
            const __bf16* src = BT + (size_t)(n0 + srow) * K + k0 + skh * 8;
            __bf16* dst = &Bs[buf][(srow >> 4) * 512 + ((srow & 15) + 16 * skh) * 16];
            copy16_g2l(src, dst);
            copy16_g2l(src + 16, dst + 8);
        }
    };

    const int nsteps = K >> 5;
    stage(0, 0);
    wait_async();
    __syncthreads();

    for (int ks = 0; ks < nsteps; ++ks) {
        const int cur = ks & 1;
        if (ks + 1 < nsteps) stage(1 - cur, (ks + 1) * 32);
        if (ks + 2 < nsteps) {   // keep k+2 flowing toward the WGP caches
            const int esz = ABF ? 2 : 4;
            __builtin_prefetch((const char*)Ap + ((size_t)(m0 + srow) * K + (ks + 2) * 32) * esz, 0, 3);
            __builtin_prefetch(BT + (size_t)(n0 + srow) * K + (ks + 2) * 32, 0, 3);
        }

        const v16bf a0 = *(const v16bf*)&As[cur][(wm * 2 + 0) * 512 + lane * 16];
        const v16bf a1 = *(const v16bf*)&As[cur][(wm * 2 + 1) * 512 + lane * 16];
        #pragma unroll
        for (int j = 0; j < 4; ++j) {
            const v16bf b = *(const v16bf*)&Bs[cur][(wn * 4 + j) * 512 + lane * 16];
            acc[0][j] = __builtin_amdgcn_wmma_f32_16x16x32_bf16(
                false, a0, false, b, (short)0, acc[0][j], false, false);
            acc[1][j] = __builtin_amdgcn_wmma_f32_16x16x32_bf16(
                false, a1, false, b, (short)0, acc[1][j], false, false);
        }
        wait_async();        // next tile's async DMA has landed
        __syncthreads();     // all waves done reading buf[cur]
    }

    // Epilogue. C/D fragment: VGPR r, lanes 0-15 -> M=r, lanes 16-31 -> M=8+r.
    const int half = lane >> 4;
    const int nl   = lane & 15;
    #pragma unroll
    for (int i = 0; i < 2; ++i) {
        #pragma unroll
        for (int j = 0; j < 4; ++j) {
            const int nc = n0 + wn * 64 + j * 16 + nl;
            #pragma unroll
            for (int r = 0; r < 8; ++r) {
                const int mr = m0 + wm * 32 + i * 16 + 8 * half + r;
                const size_t idx = (size_t)mr * Nc + nc;
                const float v = acc[i][j][r];
                if      (MODE == 0) C[idx] = v;
                else if (MODE == 1) C[idx] = v + Res[idx];
                else if (MODE == 2) C[idx] = fmaxf(v, 0.f) + Res[idx];
                else if (MODE == 3) C[idx] += v;
                else if (MODE == 4) C[idx] = (Adj[idx] != 0.f) ? v : NEG_MHA;
            }
        }
    }
}

// ---------------------------------------------------------------------------
// Row-parallel softmax family: one block of 256 threads per row of length 4096
// (16 values per lane in registers), LDS tree reductions, bf16 output so the
// downstream WMMA GEMM reads the score matrix directly as its A operand.
// ---------------------------------------------------------------------------
static __device__ __forceinline__ float blk_max(float v, float* red) {
    const int tid = threadIdx.x;
    red[tid] = v; __syncthreads();
    for (int s = 128; s > 0; s >>= 1) {
        if (tid < s) red[tid] = fmaxf(red[tid], red[tid + s]);
        __syncthreads();
    }
    const float r = red[0]; __syncthreads();
    return r;
}
static __device__ __forceinline__ float blk_sum(float v, float* red) {
    const int tid = threadIdx.x;
    red[tid] = v; __syncthreads();
    for (int s = 128; s > 0; s >>= 1) {
        if (tid < s) red[tid] += red[tid + s];
        __syncthreads();
    }
    const float r = red[0]; __syncthreads();
    return r;
}

// GAT: leaky_relu(si[i]+sj[j]), mask adj>0 -> NEG_GAT, row softmax -> bf16.
__global__ __launch_bounds__(256)
void graph_attn_softmax(const float* __restrict__ si, const float* __restrict__ sj,
                        const float* __restrict__ adj, __bf16* __restrict__ attnb)
{
    __shared__ float red[256];
    const int i = blockIdx.x, tid = threadIdx.x;
    const float sv = si[i];
    float s[16];
    float m = -3.4e38f;
    #pragma unroll
    for (int t = 0; t < 16; ++t) {
        const int j = tid + t * 256;
        float sc = sv + sj[j];
        sc = sc > 0.f ? sc : LEAKY_SLOPE * sc;
        const float a = adj[(size_t)i * NN + j];
        sc = a > 0.f ? sc : NEG_GAT;
        s[t] = sc;
        m = fmaxf(m, sc);
    }
    m = blk_max(m, red);
    float sum = 0.f;
    #pragma unroll
    for (int t = 0; t < 16; ++t) { const float e = __expf(s[t] - m); s[t] = e; sum += e; }
    sum = blk_sum(sum, red);
    const float inv = 1.f / sum;
    #pragma unroll
    for (int t = 0; t < 16; ++t) attnb[(size_t)i * NN + tid + t * 256] = f2bf(s[t] * inv);
}

// Row softmax of pre-masked f32 scores -> bf16 attention matrix.
__global__ __launch_bounds__(256)
void row_softmax(const float* __restrict__ S, __bf16* __restrict__ Sb)
{
    __shared__ float red[256];
    const int i = blockIdx.x, tid = threadIdx.x;
    float s[16];
    float m = -3.4e38f;
    #pragma unroll
    for (int t = 0; t < 16; ++t) {
        s[t] = S[(size_t)i * NN + tid + t * 256];
        m = fmaxf(m, s[t]);
    }
    m = blk_max(m, red);
    float sum = 0.f;
    #pragma unroll
    for (int t = 0; t < 16; ++t) { const float e = __expf(s[t] - m); s[t] = e; sum += e; }
    sum = blk_sum(sum, red);
    const float inv = 1.f / sum;
    #pragma unroll
    for (int t = 0; t < 16; ++t) Sb[(size_t)i * NN + tid + t * 256] = f2bf(s[t] * inv);
}

// adj2 = softmax(adj, axis=1) * adj + adj   -> bf16
__global__ __launch_bounds__(256)
void adj2_softmax(const float* __restrict__ adj, __bf16* __restrict__ Sb)
{
    __shared__ float red[256];
    const int i = blockIdx.x, tid = threadIdx.x;
    float a[16];
    float m = -3.4e38f;
    #pragma unroll
    for (int t = 0; t < 16; ++t) {
        a[t] = adj[(size_t)i * NN + tid + t * 256];
        m = fmaxf(m, a[t]);
    }
    m = blk_max(m, red);
    float sum = 0.f;
    float e[16];
    #pragma unroll
    for (int t = 0; t < 16; ++t) { e[t] = __expf(a[t] - m); sum += e[t]; }
    sum = blk_sum(sum, red);
    const float inv = 1.f / sum;
    #pragma unroll
    for (int t = 0; t < 16; ++t)
        Sb[(size_t)i * NN + tid + t * 256] = f2bf(e[t] * inv * a[t] + a[t]);
}

// si = H @ a_src, sj = H @ a_dst   (one block per row, 256-lane dot)
__global__ __launch_bounds__(256)
void si_sj_kernel(const float* __restrict__ H, const float* __restrict__ a_src,
                  const float* __restrict__ a_dst,
                  float* __restrict__ si, float* __restrict__ sj)
{
    __shared__ float red[256];
    const int i = blockIdx.x, tid = threadIdx.x;
    const float h = H[(size_t)i * FD + tid];
    const float r0 = blk_sum(h * a_src[tid], red);
    if (tid == 0) si[i] = r0;
    const float r1 = blk_sum(h * a_dst[tid], red);
    if (tid == 0) sj[i] = r1;
}

// dst[C x R] (bf16) = transpose(src[R x C]) * scale
__global__ void tconvert(const float* __restrict__ src, __bf16* __restrict__ dst,
                         int R, int C, float scale)
{
    const int idx = blockIdx.x * 256 + threadIdx.x;
    if (idx >= R * C) return;
    const int r = idx / C, c = idx % C;
    dst[(size_t)c * R + r] = f2bf(src[idx] * scale);
}

// dst (bf16) = src (f32), same layout
__global__ void convert(const float* __restrict__ src, __bf16* __restrict__ dst, int n)
{
    const int idx = blockIdx.x * 256 + threadIdx.x;
    if (idx < n) dst[idx] = f2bf(src[idx]);
}

// N2 = Zatt + leaky_relu(TMP)
__global__ void n2_kernel(const float* __restrict__ Zatt, const float* __restrict__ TMP,
                          float* __restrict__ N2, int n)
{
    const int idx = blockIdx.x * 256 + threadIdx.x;
    if (idx < n) {
        const float t = TMP[idx];
        N2[idx] = Zatt[idx] + (t > 0.f ? t : LEAKY_SLOPE * t);
    }
}

// out[i] = bias[i % FD]
__global__ void bias_init(float* __restrict__ out, const float* __restrict__ bias, int n)
{
    const int idx = blockIdx.x * 256 + threadIdx.x;
    if (idx < n) out[idx] = bias[idx & (FD - 1)];
}

// ---------------------------------------------------------------------------
// Host-side orchestration.
// ---------------------------------------------------------------------------
extern "C" void kernel_launch(void* const* d_in, const int* in_sizes, int n_in,
                              void* d_out, int out_size, void* d_ws, size_t ws_size,
                              hipStream_t stream)
{
    const float* node = (const float*)d_in[0];
    const float* adjs = (const float*)d_in[1];
    const float* Wm   = (const float*)d_in[2];
    const float* watt = (const float*)d_in[3];   // (2F,1): a_src | a_dst
    const float* wqs  = (const float*)d_in[4];
    const float* wks  = (const float*)d_in[5];
    const float* wvs  = (const float*)d_in[6];
    const float* fcm  = (const float*)d_in[7];
    const float* wat  = (const float*)d_in[8];   // (3F, F)
    const float* wgt  = (const float*)d_in[9];   // (F*E, OUT)
    const float* bias = (const float*)d_in[10];
    float* out = (float*)d_out;

    // Workspace carve-up (256B aligned).
    char* w = (char*)d_ws;
    auto alloc = [&](size_t bytes) -> char* {
        char* p = w; w += (bytes + 255) & ~(size_t)255; return p;
    };
    const size_t NNb = (size_t)NN * NN * sizeof(float);
    const size_t NNh = (size_t)NN * NN * sizeof(__bf16);
    const size_t NFb = (size_t)NN * FD * sizeof(float);
    const size_t NFh = (size_t)NN * FD * sizeof(__bf16);
    const size_t FFh = (size_t)FD * FD * sizeof(__bf16);

    float*  S    = (float*)alloc(NNb);    // f32 masked-score scratch (MHA)
    __bf16* Sb   = (__bf16*)alloc(NNh);   // bf16 attn / adj2 scratch (GEMM A operand)
    float*  H    = (float*)alloc(NFb);
    float*  Q    = (float*)alloc(NFb);
    float*  Kf   = (float*)alloc(NFb);
    float*  V    = (float*)alloc(NFb);
    float*  Zatt = (float*)alloc(NFb);
    float*  O1   = (float*)alloc(NFb);
    float*  Zat  = (float*)alloc(NFb);
    float*  TMP  = (float*)alloc(NFb);
    float*  N2   = (float*)alloc(NFb);
    float*  SUP  = (float*)alloc(NFb);
    __bf16* nodeT = (__bf16*)alloc(NFh);
    __bf16* Kb    = (__bf16*)alloc(NFh);
    __bf16* VT    = (__bf16*)alloc(NFh);
    __bf16* N2T   = (__bf16*)alloc(NFh);
    __bf16* WT    = (__bf16*)alloc(FFh);
    __bf16* WqT   = (__bf16*)alloc(FFh);
    __bf16* WkT   = (__bf16*)alloc(FFh);
    __bf16* WvT   = (__bf16*)alloc(FFh);
    __bf16* fcT   = (__bf16*)alloc(FFh);
    __bf16* WaT[3], *WeT[3];
    for (int e = 0; e < 3; ++e) { WaT[e] = (__bf16*)alloc(FFh); }
    for (int e = 0; e < 3; ++e) { WeT[e] = (__bf16*)alloc(FFh); }
    float* si = (float*)alloc((size_t)NN * sizeof(float));
    float* sj = (float*)alloc((size_t)NN * sizeof(float));

    const int cvFF = (FD * FD + 255) / 256;
    const int cvNF = (NN * FD + 255) / 256;

    // Weight transposes + bf16 conversion (B-side operands, [N x K] layout).
    tconvert<<<cvFF, 256, 0, stream>>>(Wm,  WT,  FD, FD, 1.0f);
    tconvert<<<cvFF, 256, 0, stream>>>(wqs, WqT, FD, FD, 1.0f / 16.0f); // 1/sqrt(F)
    tconvert<<<cvFF, 256, 0, stream>>>(wks, WkT, FD, FD, 1.0f);
    tconvert<<<cvFF, 256, 0, stream>>>(wvs, WvT, FD, FD, 1.0f);
    tconvert<<<cvFF, 256, 0, stream>>>(fcm, fcT, FD, FD, 1.0f);
    for (int e = 0; e < 3; ++e) {
        tconvert<<<cvFF, 256, 0, stream>>>(wat + (size_t)e * FD * FD, WaT[e], FD, FD, 1.0f);
        tconvert<<<cvFF, 256, 0, stream>>>(wgt + (size_t)e * FD * FD, WeT[e], FD, FD, 1.0f);
    }
    tconvert<<<cvNF, 256, 0, stream>>>(node, nodeT, NN, FD, 1.0f);

    #define GEMM(MODE, ABF, A_, BT_, C_, RES_, ADJ_, M_, NC_, K_) \
        gemm_bf16<MODE, ABF><<<dim3((M_) / 128, (NC_) / 128), 256, 0, stream>>>( \
            (const void*)(A_), BT_, C_, RES_, ADJ_, M_, NC_, K_)

    // Shared projections (identical across edge types): H, Q/temp, K, V.
    GEMM(0, false, node, WT,  H,  nullptr, nullptr, NN, FD, FD);
    GEMM(0, false, node, WqT, Q,  nullptr, nullptr, NN, FD, FD);
    GEMM(0, false, node, WkT, Kf, nullptr, nullptr, NN, FD, FD);
    GEMM(0, false, node, WvT, V,  nullptr, nullptr, NN, FD, FD);
    convert<<<cvNF, 256, 0, stream>>>(Kf, Kb, NN * FD);        // K rows are BT for QK^T
    tconvert<<<cvNF, 256, 0, stream>>>(V, VT, NN, FD, 1.0f);   // V^T for att@V
    si_sj_kernel<<<NN, 256, 0, stream>>>(H, watt, watt + FD, si, sj);

    bias_init<<<cvNF, 256, 0, stream>>>(out, bias, NN * FD);

    for (int e = 0; e < 3; ++e) {
        const float* adj = adjs + (size_t)e * NN * NN;

        // --- graph attention branch ---
        graph_attn_softmax<<<NN, 256, 0, stream>>>(si, sj, adj, Sb);
        GEMM(1, true, Sb, nodeT, Zatt, H, nullptr, NN, FD, NN);   // z_att = h + attn@node

        // --- MHA branch ---
        GEMM(4, false, Q, Kb, S, nullptr, adj, NN, NN, FD);       // masked (q/temp)@k^T
        row_softmax<<<NN, 256, 0, stream>>>(S, Sb);
        GEMM(0, true, Sb, VT, O1, nullptr, nullptr, NN, FD, NN);  // att@v
        GEMM(2, false, O1, fcT, Zat, node, nullptr, NN, FD, FD);  // relu(o@fc)+node

        // --- z_a = leaky(concat([z_att,z_at,node]) @ weight_at), n2 = z_att + z_a
        GEMM(0, false, Zatt, WaT[0], TMP, nullptr, nullptr, NN, FD, FD);
        GEMM(3, false, Zat,  WaT[1], TMP, nullptr, nullptr, NN, FD, FD);
        GEMM(3, false, node, WaT[2], TMP, nullptr, nullptr, NN, FD, FD);
        n2_kernel<<<cvNF, 256, 0, stream>>>(Zatt, TMP, N2, NN * FD);
        tconvert<<<cvNF, 256, 0, stream>>>(N2, N2T, NN, FD, 1.0f);

        // --- adj2 @ n2, then fold into output via this edge's weight slice ---
        adj2_softmax<<<NN, 256, 0, stream>>>(adj, Sb);
        GEMM(0, true, Sb, N2T, SUP, nullptr, nullptr, NN, FD, NN);
        GEMM(3, false, SUP, WeT[e], out, nullptr, nullptr, NN, FD, FD);
    }
    #undef GEMM
}
